// Decoder_8916352106707
// MI455X (gfx1250) — matrix operand order repaired and checked by reading, not compiled, not run on hardware
//
#include <hip/hip_runtime.h>
#include <hip/hip_bf16.h>
#include <stdint.h>

// ---------------------------------------------------------------------------
// Decoder: T=64 LSTM+attention steps, then [2048,1024]x[1024,32000] projection
// GEMMs via v_wmma_f32_16x16x32_bf16; A tiles staged with
// global_load_async_to_lds_b128 (ASYNCcnt) for the CDNA5 async path.
// ---------------------------------------------------------------------------

typedef __attribute__((ext_vector_type(16))) __bf16 v16bf_t;
typedef __attribute__((ext_vector_type(8)))  float  v8f_t;

union FragBF { v16bf_t v; uint32_t u[8]; };

__device__ __forceinline__ uint16_t f2bf(float f) {
  uint32_t u = __float_as_uint(f);
  u += 0x7FFFu + ((u >> 16) & 1u);          // round-to-nearest-even
  return (uint16_t)(u >> 16);
}
__device__ __forceinline__ float sigmf(float x) { return 1.0f / (1.0f + expf(-x)); }

__device__ __forceinline__ void store_out(float* p, float v)    { *p = v; }
__device__ __forceinline__ void store_out(uint16_t* p, float v) { *p = f2bf(v); }

// Async global->LDS 16-byte copy (CDNA5, tracked by ASYNCcnt).
__device__ __forceinline__ void async_copy_b128(const void* gptr, void* lds_ptr) {
  uint32_t lds_off = (uint32_t)(uintptr_t)lds_ptr;   // low 32 bits = LDS offset
  asm volatile("global_load_async_to_lds_b128 %0, %1, off"
               :: "v"(lds_off), "v"(gptr) : "memory");
}
__device__ __forceinline__ void async_wait0() {
  asm volatile("s_wait_asynccnt 0x0" ::: "memory");
}

// LDS A-tile row stride: 40 halves = 80 B (16B-aligned for B128 writes;
// 20-dword stride => rows 0..15 land on distinct banks mod 64).
#define AROW 40

// ---------------------------------------------------------------------------
// Generic WMMA GEMM:  C[M,N] = A[M,K](bf16) * W[N,K](bf16)^T + bias0 + bias1
// Block = 128 threads (4 waves). Block tile: (16*MT) x 64.
// grid.x = N/64, grid.y = M/(16*MT). Each wave: MT accumulators sharing one
// B fragment per k-step (MT x reuse of the weight stream).
// ---------------------------------------------------------------------------
template<int ACT, int MT, typename OutT>
__global__ __launch_bounds__(128)
void wmma_gemm_nt(const uint16_t* __restrict__ A, int lda,
                  const uint16_t* __restrict__ W, int K,
                  const float* __restrict__ bias0, const float* __restrict__ bias1,
                  OutT* __restrict__ C, int ldc)
{
  __shared__ uint16_t As[16 * MT][AROW];
  const int wave = threadIdx.x >> 5;
  const int lane = threadIdx.x & 31;
  const int row  = lane & 15;
  const int hb   = lane >> 4;                 // half-wave select
  const int m0   = blockIdx.y * (16 * MT);
  const int n0   = blockIdx.x * 64 + wave * 16;

  v8f_t acc[MT];
#pragma unroll
  for (int mt = 0; mt < MT; ++mt) acc[mt] = v8f_t{0.f,0.f,0.f,0.f,0.f,0.f,0.f,0.f};

  const uint16_t* wbase = W + (size_t)(n0 + row) * K + hb * 16;

  for (int k0 = 0; k0 < K; k0 += 32) {
    // async A tile fill: (16*MT) rows x 32 bf16 = MT*64 16-byte chunks
#pragma unroll
    for (int ch = threadIdx.x; ch < MT * 64; ch += 128) {
      int r = ch >> 2, q = (ch & 3) << 3;     // q in halves (0,8,16,24)
      async_copy_b128(A + (size_t)(m0 + r) * lda + k0 + q, &As[r][q]);
    }
    async_wait0();
    __syncthreads();

    FragBF b;
    const uint16_t* wr = wbase + k0;
    __builtin_prefetch(wr + 32, 0, 0);        // next k-block of W
#pragma unroll
    for (int v = 0; v < 8; ++v) b.u[v] = *(const uint32_t*)(wr + (v << 1));

#pragma unroll
    for (int mt = 0; mt < MT; ++mt) {
      FragBF a;
#pragma unroll
      for (int v = 0; v < 8; ++v) {           // contiguous k = 16*hb + s, matching B
        int kk = (v << 1) + (hb << 4);
        a.u[v] = *(const uint32_t*)&As[mt * 16 + row][kk];
      }
      acc[mt] = __builtin_amdgcn_wmma_f32_16x16x32_bf16(false, a.v, false, b.v,
                                                        (short)0, acc[mt], false, false);
    }
    __syncthreads();
  }

  const int n = n0 + row;
  float bias = 0.f;
  if (bias0) bias += bias0[n];
  if (bias1) bias += bias1[n];
#pragma unroll
  for (int mt = 0; mt < MT; ++mt) {
#pragma unroll
    for (int v = 0; v < 8; ++v) {             // D layout: M = v + 8*hb
      int m = m0 + mt * 16 + v + (hb << 3);
      float val = acc[mt][v] + bias;
      if (ACT == 1) val = tanhf(val);
      store_out(&C[(size_t)m * ldc + n], val);
    }
  }
}

// ---------------------------------------------------------------------------
// Fused LSTM step: gates_h = h @ W_hh^T (WMMA), + precomputed gates_x,
// then cell update. One block per 16 hidden cols; wave g handles gate g.
// Reads h(bf16) from catCur[:,0:1024], writes new h(bf16) to catNxt[:,0:1024].
// ---------------------------------------------------------------------------
__global__ __launch_bounds__(128)
void lstm_step_kernel(const uint16_t* __restrict__ catCur, // [32][2048] bf16
                      const uint16_t* __restrict__ Whh,    // [4096][1024] bf16
                      const float* __restrict__ gx,        // [32][4096] this step
                      float* __restrict__ cSt,             // [32][1024]
                      float* __restrict__ hSt,             // [32][1024]
                      uint16_t* __restrict__ catNxt)       // [32][2048] bf16
{
  __shared__ uint16_t As[32][AROW];
  __shared__ float gt[4][32][16];
  const int g    = threadIdx.x >> 5;          // gate index i,f,g,o
  const int lane = threadIdx.x & 31;
  const int row  = lane & 15;
  const int hb   = lane >> 4;
  const int jt   = blockIdx.x * 16;           // column tile within H
  const int col0 = g * 1024 + jt;

  v8f_t acc0 = {0.f,0.f,0.f,0.f,0.f,0.f,0.f,0.f};
  v8f_t acc1 = {0.f,0.f,0.f,0.f,0.f,0.f,0.f,0.f};
  const uint16_t* wbase = Whh + (size_t)(col0 + row) * 1024 + hb * 16;

  for (int k0 = 0; k0 < 1024; k0 += 32) {
    {
      // async h-tile fill: 32 rows x 32 bf16 = 128 chunks, one per thread
      int r = threadIdx.x >> 2, q = (threadIdx.x & 3) << 3;
      async_copy_b128(catCur + (size_t)r * 2048 + k0 + q, &As[r][q]);
    }
    async_wait0();
    __syncthreads();

    FragBF a0, a1, b;
#pragma unroll
    for (int v = 0; v < 8; ++v) {
      int kk = (v << 1) + (hb << 4);
      a0.u[v] = *(const uint32_t*)&As[row][kk];
      a1.u[v] = *(const uint32_t*)&As[row + 16][kk];
    }
    const uint16_t* wr = wbase + k0;
    __builtin_prefetch(wr + 32, 0, 0);
#pragma unroll
    for (int v = 0; v < 8; ++v) b.u[v] = *(const uint32_t*)(wr + (v << 1));
    acc0 = __builtin_amdgcn_wmma_f32_16x16x32_bf16(false, a0.v, false, b.v,
                                                   (short)0, acc0, false, false);
    acc1 = __builtin_amdgcn_wmma_f32_16x16x32_bf16(false, a1.v, false, b.v,
                                                   (short)0, acc1, false, false);
    __syncthreads();
  }

  const int n   = row;
  const int col = col0 + n;
#pragma unroll
  for (int v = 0; v < 8; ++v) {
    int m = v + (hb << 3);
    gt[g][m][n]      = acc0[v] + gx[(size_t)m * 4096 + col];
    gt[g][m + 16][n] = acc1[v] + gx[(size_t)(m + 16) * 4096 + col];
  }
  __syncthreads();

  for (int idx = threadIdx.x; idx < 512; idx += 128) {
    int bb = idx >> 4, j = idx & 15;
    float iv = gt[0][bb][j], fv = gt[1][bb][j], gv = gt[2][bb][j], ov = gt[3][bb][j];
    int hcol = jt + j;
    float cOld = cSt[bb * 1024 + hcol];
    float cNew = sigmf(fv) * cOld + sigmf(iv) * tanhf(gv);
    float hNew = sigmf(ov) * tanhf(cNew);
    cSt[bb * 1024 + hcol] = cNew;
    hSt[bb * 1024 + hcol] = hNew;
    catNxt[bb * 2048 + hcol] = f2bf(hNew);
  }
}

// ---------------------------------------------------------------------------
// Luong dot attention: scores->softmax->weighted sum; writes weighted (bf16)
// into catNxt[:,1024:2048]. One block per batch row, 8 waves.
// ---------------------------------------------------------------------------
__global__ __launch_bounds__(256)
void attention_kernel(const float* __restrict__ hSt,   // [32][1024]
                      const float* __restrict__ ctx,   // [32][64][1024]
                      uint16_t* __restrict__ catNxt)   // [32][2048]
{
  __shared__ float sc[64];
  const int b    = blockIdx.x;
  const int wave = threadIdx.x >> 5, lane = threadIdx.x & 31;
  const float* hv = hSt + b * 1024;
  const float* cb = ctx + (size_t)b * 64 * 1024;

  for (int s = wave; s < 64; s += 8) {
    const float* cs = cb + (size_t)s * 1024;
    float p = 0.f;
    for (int k = lane; k < 1024; k += 32) p += cs[k] * hv[k];
#pragma unroll
    for (int off = 16; off > 0; off >>= 1) p += __shfl_xor(p, off, 32);
    if (lane == 0) sc[s] = p;
  }
  __syncthreads();
  if (wave == 0) {                            // softmax over 64 by one wave
    float v0 = sc[lane], v1 = sc[lane + 32];
    float mx = fmaxf(v0, v1);
#pragma unroll
    for (int off = 16; off > 0; off >>= 1) mx = fmaxf(mx, __shfl_xor(mx, off, 32));
    float e0 = expf(v0 - mx), e1 = expf(v1 - mx);
    float ss = e0 + e1;
#pragma unroll
    for (int off = 16; off > 0; off >>= 1) ss += __shfl_xor(ss, off, 32);
    sc[lane] = e0 / ss; sc[lane + 32] = e1 / ss;
  }
  __syncthreads();
  for (int k = threadIdx.x; k < 1024; k += 256) {
    float acc = 0.f;
    for (int s = 0; s < 64; ++s) acc += sc[s] * cb[(size_t)s * 1024 + k];
    catNxt[b * 2048 + 1024 + k] = f2bf(acc);
  }
}

// ---------------------------------------------------------------------------
// In-place log_softmax over V per row; one block (8 waves) per row.
// ---------------------------------------------------------------------------
__global__ __launch_bounds__(256)
void logsoftmax_kernel(float* __restrict__ logits, int Vv)
{
  __shared__ float red[8];
  float* rowp = logits + (size_t)blockIdx.x * Vv;
  const int wave = threadIdx.x >> 5, lane = threadIdx.x & 31;

  float mx = -3.4e38f;
  for (int i = threadIdx.x; i < Vv; i += 256) mx = fmaxf(mx, rowp[i]);
#pragma unroll
  for (int off = 16; off > 0; off >>= 1) mx = fmaxf(mx, __shfl_xor(mx, off, 32));
  if (lane == 0) red[wave] = mx;
  __syncthreads();
  float m = red[0];
#pragma unroll
  for (int i = 1; i < 8; ++i) m = fmaxf(m, red[i]);
  __syncthreads();

  float s = 0.f;
  for (int i = threadIdx.x; i < Vv; i += 256) s += expf(rowp[i] - m);
#pragma unroll
  for (int off = 16; off > 0; off >>= 1) s += __shfl_xor(s, off, 32);
  if (lane == 0) red[wave] = s;
  __syncthreads();
  float tot = 0.f;
#pragma unroll
  for (int i = 0; i < 8; ++i) tot += red[i];
  float lse = m + logf(tot);
  for (int i = threadIdx.x; i < Vv; i += 256) rowp[i] -= lse;
}

// ---------------------------------------------------------------------------
// Small utility kernels
// ---------------------------------------------------------------------------
__global__ void f32_to_bf16_kernel(const float* __restrict__ src,
                                   uint16_t* __restrict__ dst, size_t n) {
  size_t idx = (size_t)blockIdx.x * blockDim.x + threadIdx.x;
  if (idx < n) dst[idx] = f2bf(src[idx]);
}

__global__ void gather_embed_kernel(const int* __restrict__ xseq,
                                    const float* __restrict__ emb,
                                    uint16_t* __restrict__ e16, int total) {
  int idx = blockIdx.x * blockDim.x + threadIdx.x;
  if (idx >= total) return;
  int r = idx >> 9, k = idx & 511;            // E = 512
  int tok = xseq[r];
  e16[idx] = f2bf(emb[(size_t)tok * 512 + k]);
}

__global__ void init_state_kernel(const float* __restrict__ h,
                                  const float* __restrict__ c,
                                  float* __restrict__ hSt, float* __restrict__ cSt,
                                  uint16_t* __restrict__ cat0) {
  int idx = blockIdx.x * blockDim.x + threadIdx.x;
  if (idx >= 32 * 1024) return;
  hSt[idx] = h[idx]; cSt[idx] = c[idx];
  int bb = idx >> 10, k = idx & 1023;
  cat0[bb * 2048 + k] = f2bf(h[idx]);
}

__global__ void copy_hc_kernel(const float* __restrict__ hSt,
                               const float* __restrict__ cSt,
                               float* __restrict__ out) {
  int idx = blockIdx.x * blockDim.x + threadIdx.x;
  if (idx >= 2 * 32768) return;
  out[idx] = (idx < 32768) ? hSt[idx] : cSt[idx - 32768];
}

// ---------------------------------------------------------------------------
// Host orchestration
// ---------------------------------------------------------------------------
extern "C" void kernel_launch(void* const* d_in, const int* in_sizes, int n_in,
                              void* d_out, int out_size, void* d_ws, size_t ws_size,
                              hipStream_t stream) {
  (void)in_sizes; (void)n_in; (void)out_size; (void)ws_size;
  const int*   x_seq = (const int*)  d_in[0];
  const float* h_in  = (const float*)d_in[1];
  const float* c_in  = (const float*)d_in[2];
  const float* ctx   = (const float*)d_in[3];
  const float* emb   = (const float*)d_in[4];
  const float* W_ih  = (const float*)d_in[5];
  const float* W_hh  = (const float*)d_in[6];
  const float* b_ih  = (const float*)d_in[7];
  const float* b_hh  = (const float*)d_in[8];
  const float* W_a   = (const float*)d_in[9];
  const float* b_a   = (const float*)d_in[10];
  const float* W_o   = (const float*)d_in[11];
  const float* b_o   = (const float*)d_in[12];

  char* ws = (char*)d_ws;
  size_t off = 0;
  auto alloc = [&](size_t bytes) -> void* {
    void* p = ws + off;
    off += (bytes + 255) & ~(size_t)255;
    return p;
  };
  uint16_t* wih16  = (uint16_t*)alloc(4096ull * 512 * 2);
  uint16_t* whh16  = (uint16_t*)alloc(4096ull * 1024 * 2);
  uint16_t* wa16   = (uint16_t*)alloc(1024ull * 2048 * 2);
  uint16_t* wo16   = (uint16_t*)alloc(32000ull * 1024 * 2);
  uint16_t* e16    = (uint16_t*)alloc(2048ull * 512 * 2);
  float*    gatesx = (float*)   alloc(2048ull * 4096 * 4);
  uint16_t* cat0   = (uint16_t*)alloc(32ull * 2048 * 2);
  uint16_t* cat1   = (uint16_t*)alloc(32ull * 2048 * 2);
  float*    hSt    = (float*)   alloc(32ull * 1024 * 4);
  float*    cSt    = (float*)   alloc(32ull * 1024 * 4);
  uint16_t* oseq16 = (uint16_t*)alloc(2048ull * 1024 * 2);

  // 1) weights -> bf16
  f32_to_bf16_kernel<<<(4096 * 512 + 255) / 256, 256, 0, stream>>>(W_ih, wih16, 4096ull * 512);
  f32_to_bf16_kernel<<<(4096 * 1024 + 255) / 256, 256, 0, stream>>>(W_hh, whh16, 4096ull * 1024);
  f32_to_bf16_kernel<<<(1024 * 2048 + 255) / 256, 256, 0, stream>>>(W_a, wa16, 1024ull * 2048);
  f32_to_bf16_kernel<<<(32768000 + 255) / 256, 256, 0, stream>>>(W_o, wo16, 32000ull * 1024);

  // 2) embedding gather -> bf16
  gather_embed_kernel<<<(2048 * 512 + 255) / 256, 256, 0, stream>>>(x_seq, emb, e16, 2048 * 512);

  // 3) hoisted input-gate GEMM for all T: gates_x = e @ W_ih^T + b_ih + b_hh
  wmma_gemm_nt<0, 4, float><<<dim3(4096 / 64, 2048 / 64), 128, 0, stream>>>(
      e16, 512, wih16, 512, b_ih, b_hh, gatesx, 4096);

  // 4) initial state
  init_state_kernel<<<(32768 + 255) / 256, 256, 0, stream>>>(h_in, c_in, hSt, cSt, cat0);

  // 5) recurrence
  uint16_t* cur = cat0;
  uint16_t* nxt = cat1;
  for (int t = 0; t < 64; ++t) {
    lstm_step_kernel<<<64, 128, 0, stream>>>(cur, whh16, gatesx + (size_t)t * 32 * 4096,
                                             cSt, hSt, nxt);
    attention_kernel<<<32, 256, 0, stream>>>(hSt, ctx, nxt);
    wmma_gemm_nt<1, 2, uint16_t><<<dim3(1024 / 64, 1), 128, 0, stream>>>(
        nxt, 2048, wa16, 2048, b_a, nullptr, oseq16 + (size_t)t * 32 * 1024, 1024);
    uint16_t* tmp = cur; cur = nxt; nxt = tmp;
  }

  // 6) output projection straight into d_out, then in-place log_softmax
  float* logits = (float*)d_out;
  wmma_gemm_nt<0, 4, float><<<dim3(32000 / 64, 2048 / 64), 128, 0, stream>>>(
      oseq16, 1024, wo16, 1024, b_o, nullptr, logits, 32000);
  logsoftmax_kernel<<<2048, 256, 0, stream>>>(logits, 32000);

  // 7) final h, c appended after logits
  copy_hc_kernel<<<(65536 + 255) / 256, 256, 0, stream>>>(hSt, cSt, logits + 2048ull * 32000);
}
